// LineGraphHiddenLayer_19069654794984
// MI455X (gfx1250) — matrix operand order repaired and checked by reading, not compiled, not run on hardware
//
#include <hip/hip_runtime.h>
#include <hip/hip_bf16.h>
#include <math.h>
#include <stdint.h>

#define H 256
#define H4 1024
#define LAYERS 2
#define KHOPS 4

typedef __attribute__((ext_vector_type(16))) __bf16 v16bf;
typedef __attribute__((ext_vector_type(8)))  float  v8f;

static inline int ceil_div(int a, int b) { return (a + b - 1) / b; }

__device__ __forceinline__ float gelu_exact(float v) {
    // exact GELU: 0.5*x*(1+erf(x/sqrt(2)))
    return 0.5f * v * (1.0f + erff(v * 0.70710678118654752f));
}

// Flat->LDS aperture mapping uses addr[31:0] as the DS byte address, so the
// low 32 bits of a generic pointer to __shared__ memory are the LDS offset.
__device__ __forceinline__ uint32_t lds_addr32(const void* p) {
    return (uint32_t)(uintptr_t)p;
}

// CDNA5 async DMA: copy 16B global -> LDS per active lane (ASYNCcnt tracked).
__device__ __forceinline__ void async_lds_b128(uint32_t lds_off, const void* gptr) {
    asm volatile("global_load_async_to_lds_b128 %0, %1, off"
                 :: "v"(lds_off), "v"((unsigned long long)(uintptr_t)gptr)
                 : "memory");
}
__device__ __forceinline__ void wait_asynccnt0() {
    asm volatile("s_wait_asynccnt 0" ::: "memory");
}

// ---------------------------------------------------------------------------
// Weight prep: out[n][k] (bf16) = in[k][n] (f32); K,N multiples of 32
// ---------------------------------------------------------------------------
__global__ __launch_bounds__(256) void convT_kernel(__bf16* __restrict__ out,
                                                    const float* __restrict__ in,
                                                    int K, int N) {
    __shared__ float t[32][33];
    int nb = blockIdx.x * 32, kb = blockIdx.y * 32;
    int tx = threadIdx.x & 31, ty = threadIdx.x >> 5;   // 32 x 8
    #pragma unroll
    for (int i = 0; i < 32; i += 8)
        t[ty + i][tx] = in[(long)(kb + ty + i) * N + nb + tx];
    __syncthreads();
    #pragma unroll
    for (int i = 0; i < 32; i += 8)
        out[(long)(nb + ty + i) * K + kb + tx] = (__bf16)t[tx][ty + i];
}

// ---------------------------------------------------------------------------
// lg_x = rel_embed[edge_feat]   (row gather, float4 vectorized)
// ---------------------------------------------------------------------------
__global__ void gather_rows_kernel(float* __restrict__ out,
                                   const float* __restrict__ table,
                                   const int* __restrict__ idx, int nRows) {
    int t = blockIdx.x * blockDim.x + threadIdx.x;
    int total = nRows * (H / 4);
    if (t >= total) return;
    int e = t / (H / 4), c = t % (H / 4);
    const float4* s = (const float4*)(table + (long)idx[e] * H);
    float4* d = (float4*)(out + (long)e * H);
    d[c] = s[c];
}

// fused_edges[e] = x[src[e]] + x[dst[e]]
__global__ void gather_add2_kernel(float* __restrict__ out,
                                   const float* __restrict__ rows,
                                   const int* __restrict__ ia,
                                   const int* __restrict__ ib, int nRows) {
    int t = blockIdx.x * blockDim.x + threadIdx.x;
    int total = nRows * (H / 4);
    if (t >= total) return;
    int e = t / (H / 4), c = t % (H / 4);
    const float4 a = ((const float4*)(rows + (long)ia[e] * H))[c];
    const float4 b = ((const float4*)(rows + (long)ib[e] * H))[c];
    float4 r;
    r.x = a.x + b.x; r.y = a.y + b.y; r.z = a.z + b.z; r.w = a.w + b.w;
    ((float4*)(out + (long)e * H))[c] = r;
}

// fused_nodes[ia[e]] += lg[e]; fused_nodes[ib[e]] += lg[e]
__global__ void scatter_add2_kernel(float* __restrict__ out,
                                    const float* __restrict__ rows,
                                    const int* __restrict__ ia,
                                    const int* __restrict__ ib, int nRows) {
    int t = blockIdx.x * blockDim.x + threadIdx.x;
    int total = nRows * (H / 4);
    if (t >= total) return;
    int e = t / (H / 4), c = t % (H / 4);
    const float4 v = ((const float4*)(rows + (long)e * H))[c];
    float* oa = out + (long)ia[e] * H + c * 4;
    float* ob = out + (long)ib[e] * H + c * 4;
    atomicAdd(oa + 0, v.x); atomicAdd(oa + 1, v.y);
    atomicAdd(oa + 2, v.z); atomicAdd(oa + 3, v.w);
    atomicAdd(ob + 0, v.x); atomicAdd(ob + 1, v.y);
    atomicAdd(ob + 2, v.z); atomicAdd(ob + 3, v.w);
}

// segment_sum: out[dst[e]] += in[src[e]]
__global__ void scatter_seg_kernel(float* __restrict__ out,
                                   const float* __restrict__ in,
                                   const int* __restrict__ src,
                                   const int* __restrict__ dst, int nRows) {
    int t = blockIdx.x * blockDim.x + threadIdx.x;
    int total = nRows * (H / 4);
    if (t >= total) return;
    int e = t / (H / 4), c = t % (H / 4);
    const float4 v = ((const float4*)(in + (long)src[e] * H))[c];
    float* o = out + (long)dst[e] * H + c * 4;
    atomicAdd(o + 0, v.x); atomicAdd(o + 1, v.y);
    atomicAdd(o + 2, v.z); atomicAdd(o + 3, v.w);
}

// ---------------------------------------------------------------------------
// out = LayerNorm(a + (GELU_B ? gelu(b) : b)) * g + beta    (H=256 per row)
// ---------------------------------------------------------------------------
template <bool GELU_B>
__global__ void residual_ln_kernel(float* __restrict__ out,
                                   const float* __restrict__ a,
                                   const float* __restrict__ b,
                                   const float* __restrict__ g,
                                   const float* __restrict__ beta) {
    int row = blockIdx.x;
    int h = threadIdx.x;                 // 0..255
    long idx = (long)row * H + h;
    float bv = b[idx];
    if (GELU_B) bv = gelu_exact(bv);
    float v = a[idx] + bv;

    float s = v, q = v * v;
    #pragma unroll
    for (int off = 16; off > 0; off >>= 1) {
        s += __shfl_xor(s, off, 32);
        q += __shfl_xor(q, off, 32);
    }
    __shared__ float ss[8], qq[8];
    int wid = h >> 5, lane = h & 31;
    if (lane == 0) { ss[wid] = s; qq[wid] = q; }
    __syncthreads();
    float S = 0.f, Q = 0.f;
    #pragma unroll
    for (int i = 0; i < 8; ++i) { S += ss[i]; Q += qq[i]; }
    float mean = S * (1.0f / H);
    float var = Q * (1.0f / H) - mean * mean;
    out[idx] = (v - mean) * rsqrtf(var + 1e-5f) * g[h] + beta[h];
}

// ---------------------------------------------------------------------------
// C[M,N] (+)= act( A[M,K] @ Wt[N,K]^T + bias[N] ) via v_wmma_f32_16x16x32_bf16
// block = 256 threads = 8 waves; block tile 128x64; wave tile 16x64 (4 WMMA,
// one A fragment reused 4x); K stepped by 32 through LDS-staged bf16 tiles.
// Wt is pre-converted bf16, pre-transposed [N][K]; its tile is staged with
// GLOBAL_LOAD_ASYNC_TO_LDS_B128 (ASYNCcnt), as is A when already bf16.
// ---------------------------------------------------------------------------
template <bool A_BF16, bool ACC, bool GELU_BF16_OUT>
__global__ __launch_bounds__(256) void
gemm_wmma_kernel(const void* __restrict__ Av, const __bf16* __restrict__ Wt,
                 const float* __restrict__ bias, void* __restrict__ Cv,
                 int M, int K, int N) {
    __shared__ __align__(16) __bf16 aT[128][48];
    __shared__ __align__(16) __bf16 bT[64][48];

    const int tid = threadIdx.x;
    const int wave = tid >> 5;
    const int lane = tid & 31;
    const int m0 = blockIdx.y * 128;
    const int n0 = blockIdx.x * 64;

    v8f c[4];
    #pragma unroll
    for (int s = 0; s < 4; ++s) c[s] = (v8f){0.f,0.f,0.f,0.f,0.f,0.f,0.f,0.f};

    const int arow = tid >> 1;          // 0..127
    const int acol = (tid & 1) * 16;    // 0,16
    const int brow = tid >> 2;          // 0..63
    const int bcol = (tid & 3) * 8;     // 0,8,16,24

    const int gm = m0 + arow;
    const bool aOK = (gm < M);
    if (!aOK) {
        // OOB rows written only by their owner: zero once, stays zero
        #pragma unroll
        for (int j = 0; j < 16; ++j) aT[arow][acol + j] = (__bf16)0.0f;
    }

    const uint32_t ldsB  = lds_addr32(&bT[brow][bcol]);
    const uint32_t ldsA0 = lds_addr32(&aT[arow][acol]);
    const uint32_t ldsA1 = lds_addr32(&aT[arow][acol + 8]);
    const __bf16* wB = Wt + (long)(n0 + brow) * K + bcol;

    for (int kk = 0; kk < K; kk += 32) {
        // stage B tile (64 n-rows x 32 k): async DMA, 16B per lane
        async_lds_b128(ldsB, wB + kk);
        // stage A tile (128 x 32)
        if (aOK) {
            if constexpr (A_BF16) {
                const __bf16* ap = (const __bf16*)Av + (long)gm * K + kk + acol;
                async_lds_b128(ldsA0, ap);
                async_lds_b128(ldsA1, ap + 8);
            } else {
                const float* ap = (const float*)Av + (long)gm * K + kk + acol;
                __builtin_prefetch(ap + 32, 0, 3);       // next k-slab
                #pragma unroll
                for (int j = 0; j < 16; ++j) aT[arow][acol + j] = (__bf16)ap[j];
            }
        }
        wait_asynccnt0();
        __syncthreads();

        // A fragment (ISA 16-bit A 16x32 layout): lane>>4 selects K-half
        v16bf af;
        {
            int r = wave * 16 + (lane & 15);
            int hk = (lane >> 4) * 8;
            #pragma unroll
            for (int j = 0; j < 8; ++j) {
                af[j]     = aT[r][hk + j];
                af[j + 8] = aT[r][16 + hk + j];
            }
        }
        // 4 N-subtiles, one WMMA each, reusing af
        #pragma unroll
        for (int s = 0; s < 4; ++s) {
            v16bf bfr;
            int n = s * 16 + (lane & 15);
            int kb = (lane >> 4) * 8;
            #pragma unroll
            for (int j = 0; j < 16; ++j) bfr[j] = bT[n][kb + j + (j >= 8 ? 8 : 0)];
            c[s] = __builtin_amdgcn_wmma_f32_16x16x32_bf16(false, af, false, bfr,
                                                           (short)0, c[s], false, false);
        }
        __syncthreads();
    }

    // epilogue: vgpr i -> row (lane>>4)*8+i, col lane&15 (per 16x16 subtile)
    const int rbase = m0 + wave * 16 + (lane >> 4) * 8;
    #pragma unroll
    for (int s = 0; s < 4; ++s) {
        int gn = n0 + s * 16 + (lane & 15);
        float bb = bias[gn];
        #pragma unroll
        for (int i = 0; i < 8; ++i) {
            int gmo = rbase + i;
            if (gmo < M) {
                long idx = (long)gmo * N + gn;
                float v = c[s][i] + bb;
                if constexpr (GELU_BF16_OUT) {
                    ((__bf16*)Cv)[idx] = (__bf16)gelu_exact(v);
                } else {
                    float* O = (float*)Cv;
                    if (ACC) v += O[idx];
                    O[idx] = v;
                }
            }
        }
    }
}

// ---------------------------------------------------------------------------
// host-side orchestration
// ---------------------------------------------------------------------------
enum GemmKind { G_F32OUT, G_F32ACC, G_GELU_BF16OUT, G_ABF16_F32OUT };

static void launch_gemm(GemmKind kind, const void* A, const __bf16* Wt,
                        const float* bias, void* C, int M, int K, int N,
                        hipStream_t s) {
    dim3 grid(N / 64, (unsigned)ceil_div(M, 128));
    dim3 block(256);
    switch (kind) {
    case G_F32OUT:
        gemm_wmma_kernel<false, false, false><<<grid, block, 0, s>>>(A, Wt, bias, C, M, K, N);
        break;
    case G_F32ACC:
        gemm_wmma_kernel<false, true, false><<<grid, block, 0, s>>>(A, Wt, bias, C, M, K, N);
        break;
    case G_GELU_BF16OUT:
        gemm_wmma_kernel<false, false, true><<<grid, block, 0, s>>>(A, Wt, bias, C, M, K, N);
        break;
    case G_ABF16_F32OUT:
        gemm_wmma_kernel<true, false, false><<<grid, block, 0, s>>>(A, Wt, bias, C, M, K, N);
        break;
    }
}

struct CoreParams {
    const __bf16 *wsT, *wkT, *wfT, *w1T, *w2T;     // pre-transposed bf16 [N][K]
    const float *bs, *bk, *bfu, *b1, *b2, *g1, *c1, *g2, *c2;
};

static void run_core(const float* in, const float* fused, int M,
                     const int* src, const int* dst, int nEdge,
                     float* out, float* acc, float* out1, float* zA, float* zB,
                     __bf16* ffh, const CoreParams& P, hipStream_t s) {
    // acc = in @ ws + bs
    launch_gemm(G_F32OUT, in, P.wsT, P.bs, acc, M, H, H, s);
    // k-hop: z_{i+1} = segment_sum(z_i[src], dst); acc += z @ wk_i + bk_i
    const float* zin = in;
    float* bufs[2] = {zA, zB};
    for (int i = 0; i < KHOPS; ++i) {
        float* zo = bufs[i & 1];
        hipMemsetAsync(zo, 0, (size_t)M * H * sizeof(float), s);
        int total = nEdge * (H / 4);
        scatter_seg_kernel<<<ceil_div(total, 256), 256, 0, s>>>(zo, zin, src, dst, nEdge);
        launch_gemm(G_F32ACC, zo, P.wkT + (size_t)i * H * H, P.bk + (size_t)i * H,
                    acc, M, H, H, s);
        zin = zo;
    }
    // acc += fused @ wf + bf
    launch_gemm(G_F32ACC, fused, P.wfT, P.bfu, acc, M, H, H, s);
    // out1 = LN(in + gelu(acc))
    residual_ln_kernel<true><<<M, 256, 0, s>>>(out1, in, acc, P.g1, P.c1);
    // ffh(bf16) = gelu(out1 @ w1 + b1);  acc = ffh @ w2 + b2
    launch_gemm(G_GELU_BF16OUT, out1, P.w1T, P.b1, ffh, M, H, H4, s);
    launch_gemm(G_ABF16_F32OUT, ffh, P.w2T, P.b2, acc, M, H4, H, s);
    // out = LN(out1 + acc)
    residual_ln_kernel<false><<<M, 256, 0, s>>>(out, out1, acc, P.g2, P.c2);
}

extern "C" void kernel_launch(void* const* d_in, const int* in_sizes, int n_in,
                              void* d_out, int out_size, void* d_ws, size_t ws_size,
                              hipStream_t stream) {
    const float* x_in  = (const float*)d_in[0];
    const int*   efeat = (const int*)d_in[1];
    const int*   eg    = (const int*)d_in[2];
    const int*   elg   = (const int*)d_in[3];
    const float* rel   = (const float*)d_in[4];

    const int nN  = in_sizes[0] / H;
    const int nE  = in_sizes[1];
    const int nLG = in_sizes[3] / 2;
    const int* srcg = eg;        const int* dstg = eg + nE;
    const int* srcl = elg;       const int* dstl = elg + nLG;

    // ---- workspace carve: f32 arena then bf16 arena -----------------------
    float* w = (float*)d_ws;
    auto take = [&](size_t n) { float* q = w; w += n; return q; };
    float* lgA    = take((size_t)nE * H);
    float* lgB    = take((size_t)nE * H);
    float* xA     = take((size_t)nN * H);
    float* xB     = take((size_t)nN * H);
    float* fusedN = take((size_t)nN * H);
    float* fusedE = take((size_t)nE * H);
    float* acc    = take((size_t)nE * H);
    float* out1   = take((size_t)nE * H);
    float* zA     = take((size_t)nE * H);
    float* zB     = take((size_t)nE * H);
    const size_t WT_PER_LC = (size_t)6 * H * H + 2 * (size_t)H * H4;  // elems
    __bf16* wt    = (__bf16*)w;                      // 4 * WT_PER_LC bf16
    __bf16* ffh   = wt + 4 * WT_PER_LC;              // nE * H4 bf16
    (void)ws_size; (void)n_in; (void)out_size;

    // ---- per-call weight convert+transpose to bf16 [N][K] -----------------
    for (int lc = 0; lc < 4; ++lc) {
        __bf16* base = wt + (size_t)lc * WT_PER_LC;
        const float* wself = (const float*)d_in[5]  + (size_t)lc * H * H;
        const float* wkhop = (const float*)d_in[7]  + (size_t)lc * KHOPS * H * H;
        const float* wfuse = (const float*)d_in[9]  + (size_t)lc * H * H;
        const float* wff1  = (const float*)d_in[11] + (size_t)lc * H * H4;
        const float* wff2  = (const float*)d_in[13] + (size_t)lc * H4 * H;
        dim3 b(256);
        convT_kernel<<<dim3(H / 32, H / 32), b, 0, stream>>>(base, wself, H, H);
        for (int i = 0; i < KHOPS; ++i)
            convT_kernel<<<dim3(H / 32, H / 32), b, 0, stream>>>(
                base + (size_t)(1 + i) * H * H, wkhop + (size_t)i * H * H, H, H);
        convT_kernel<<<dim3(H / 32, H / 32), b, 0, stream>>>(base + (size_t)5 * H * H,
                                                             wfuse, H, H);
        convT_kernel<<<dim3(H4 / 32, H / 32), b, 0, stream>>>(base + (size_t)6 * H * H,
                                                              wff1, H, H4);
        convT_kernel<<<dim3(H / 32, H4 / 32), b, 0, stream>>>(
            base + (size_t)6 * H * H + (size_t)H * H4, wff2, H4, H);
    }

    auto P = [&](int l, int c) {
        size_t lc = (size_t)(l * 2 + c);
        __bf16* base = wt + lc * WT_PER_LC;
        CoreParams p;
        p.wsT = base;
        p.wkT = base + (size_t)1 * H * H;
        p.wfT = base + (size_t)5 * H * H;
        p.w1T = base + (size_t)6 * H * H;
        p.w2T = base + (size_t)6 * H * H + (size_t)H * H4;
        p.bs  = (const float*)d_in[6]  + lc * H;
        p.bk  = (const float*)d_in[8]  + lc * KHOPS * H;
        p.bfu = (const float*)d_in[10] + lc * H;
        p.b1  = (const float*)d_in[12] + lc * H4;
        p.b2  = (const float*)d_in[14] + lc * H;
        p.g1  = (const float*)d_in[15] + lc * H;
        p.c1  = (const float*)d_in[16] + lc * H;
        p.g2  = (const float*)d_in[17] + lc * H;
        p.c2  = (const float*)d_in[18] + lc * H;
        return p;
    };

    float* outX  = (float*)d_out;
    float* outLG = (float*)d_out + (size_t)nN * H;

    // lg_x = rel_embed[edge_feat]
    {
        int total = nE * (H / 4);
        gather_rows_kernel<<<ceil_div(total, 256), 256, 0, stream>>>(lgA, rel, efeat, nE);
    }

    const float* xcur = x_in;
    const float* lgcur = lgA;
    for (int l = 0; l < LAYERS; ++l) {
        // fused_nodes = segsum(lg, src_g) + segsum(lg, dst_g); fused_edges gather
        hipMemsetAsync(fusedN, 0, (size_t)nN * H * sizeof(float), stream);
        {
            int total = nE * (H / 4);
            scatter_add2_kernel<<<ceil_div(total, 256), 256, 0, stream>>>(
                fusedN, lgcur, srcg, dstg, nE);
            gather_add2_kernel<<<ceil_div(total, 256), 256, 0, stream>>>(
                fusedE, xcur, srcg, dstg, nE);
        }

        const bool last = (l == LAYERS - 1);
        float* xout  = last ? outX  : ((l & 1) ? xB : xA);
        float* lgout = last ? outLG : ((l & 1) ? lgA : lgB);

        run_core(xcur,  fusedN, nN, srcg, dstg, nE,  xout,  acc, out1, zA, zB, ffh,
                 P(l, 0), stream);
        run_core(lgcur, fusedE, nE, srcl, dstl, nLG, lgout, acc, out1, zA, zB, ffh,
                 P(l, 1), stream);

        xcur = xout;
        lgcur = lgout;
    }
}
